// LoraLinear_62156766708153
// MI455X (gfx1250) — compile-verified, matrix operand order checked by reading, not run on hardware
//
#include <hip/hip_runtime.h>
#include <hip/hip_bf16.h>
#include <cstdint>

typedef __attribute__((ext_vector_type(16))) __bf16 v16bf;
typedef __attribute__((ext_vector_type(2)))  __bf16 v2bf;
typedef __attribute__((ext_vector_type(8)))  float  v8f;
typedef __attribute__((vector_size(16)))     int    i32x4;   // matches builtin proto

#define DEVINL __device__ __forceinline__
#define AS1 __attribute__((address_space(1)))
#define AS3 __attribute__((address_space(3)))

#if defined(__has_builtin)
#  if __has_builtin(__builtin_amdgcn_global_load_async_to_lds_b128)
#    define HAVE_ASYNC_LDS 1
#  endif
#endif
#ifndef HAVE_ASYNC_LDS
#  define HAVE_ASYNC_LDS 0
#endif

constexpr int IN_FEATURES = 4096;
constexpr int LORA_R      = 128;
constexpr int OUT_DIM     = 4096;
constexpr int BSZ         = 8;
constexpr int SEQ         = 2048;
constexpr int M_TOTAL     = BSZ * SEQ;     // 16384
constexpr float SCALE     = 2.0f;          // alpha / r

// padded K-strides (halves): conflict-free ds_load_b128 fragment access
constexpr int KP1 = 40;    // kernel1 tiles: K=32 data + 8 pad  (80B row)
constexpr int KP2 = 136;   // kernel2 tiles: K=128 data + 8 pad (272B row)

// ---- helpers -------------------------------------------------------------

// hardware fp32->bf16 (backend emits v_cvt_*bf16_f32), packed into a dword
DEVINL unsigned pack2_bf16(float a, float b) {
  v2bf v;
  v[0] = (__bf16)a;
  v[1] = (__bf16)b;
  return __builtin_bit_cast(unsigned, v);
}

// One 16x32 bf16 fragment slice for this lane from an LDS tile row:
// chunk0 = K[kb..kb+7], chunk1 = K[kb+16..kb+23] (ISA 16-bit A/B layout)
DEVINL v16bf load_frag(const unsigned short* p) {
  union { uint4 u[2]; v16bf v; } f;
  f.u[0] = *(const uint4*)(p);
  f.u[1] = *(const uint4*)(p + 16);
  return f.v;
}

DEVINL void wait_async_then_barrier() {
#if HAVE_ASYNC_LDS
#  if __has_builtin(__builtin_amdgcn_s_wait_asynccnt)
  __builtin_amdgcn_s_wait_asynccnt(0);
#  else
  asm volatile("s_wait_asynccnt 0x0" ::: "memory");
#  endif
#endif
  __syncthreads();
}

// ---- stage 1: down[m, r] = x[m, :] . W_down[r, :]   (bf16 out) ------------
// grid.x = M_TOTAL/128, block = 256 (8 wave32). WG tile 128(M) x 128(N), K step 32.

__global__ __launch_bounds__(256) void lora_down_wmma(
    const float* __restrict__ x, const float* __restrict__ Wd,
    __bf16* __restrict__ down)
{
  __shared__ unsigned short As[2][128 * KP1];  // x tile [m][k] bf16, double buffered
  __shared__ unsigned short Bs[2][128 * KP1];  // W_down tile [n][k] bf16 (col-major B)

  const int tid   = threadIdx.x;
  const int lane  = tid & 31;
  const int wid   = tid >> 5;
  const int mtile = blockIdx.x;

  // staging role: 2 threads per row, 16 floats (64B) each
  const int srow = tid >> 1;
  const int sseg = tid & 1;
  const float* xrow = x  + (size_t)(mtile * 128 + srow) * IN_FEATURES + sseg * 16;
  const float* wrow = Wd + (size_t)srow * IN_FEATURES + sseg * 16;

  v8f acc[8] = {};

  // prologue: stage K-tile 0 into buffer 0
  {
    const float4* xa = (const float4*)xrow;
    const float4* wa = (const float4*)wrow;
    unsigned xd[8], wd[8];
    #pragma unroll
    for (int i = 0; i < 4; ++i) {
      float4 xv = xa[i], wv = wa[i];
      xd[2*i+0] = pack2_bf16(xv.x, xv.y);
      xd[2*i+1] = pack2_bf16(xv.z, xv.w);
      wd[2*i+0] = pack2_bf16(wv.x, wv.y);
      wd[2*i+1] = pack2_bf16(wv.z, wv.w);
    }
    uint4* ad = (uint4*)&As[0][srow * KP1 + sseg * 16];
    uint4* bd = (uint4*)&Bs[0][srow * KP1 + sseg * 16];
    ad[0] = *(uint4*)&xd[0]; ad[1] = *(uint4*)&xd[4];
    bd[0] = *(uint4*)&wd[0]; bd[1] = *(uint4*)&wd[4];
  }
  __syncthreads();

  const int kb   = (lane < 16) ? 0 : 8;
  const int arow = wid * 16 + (lane & 15);

  constexpr int KT = IN_FEATURES / 32;   // 128 K-steps
  for (int kt = 0; kt < KT; ++kt) {
    const int cur = kt & 1;
    const int nxt = cur ^ 1;

    // register-prefetch next K-tile (overlaps with WMMA below)
    unsigned xd[8], wd[8];
    if (kt + 1 < KT) {
      const float4* xa = (const float4*)(xrow + (size_t)(kt + 1) * 32);
      const float4* wa = (const float4*)(wrow + (size_t)(kt + 1) * 32);
      #pragma unroll
      for (int i = 0; i < 4; ++i) {
        float4 xv = xa[i], wv = wa[i];
        xd[2*i+0] = pack2_bf16(xv.x, xv.y);
        xd[2*i+1] = pack2_bf16(xv.z, xv.w);
        wd[2*i+0] = pack2_bf16(wv.x, wv.y);
        wd[2*i+1] = pack2_bf16(wv.z, wv.w);
      }
    }
    if (kt + 2 < KT)   // L2 prefetch two tiles ahead (global_prefetch_b8)
      __builtin_prefetch((const void*)(xrow + (size_t)(kt + 2) * 32), 0, 1);

    // load A fragment + all 8 B fragments first, then issue 8 WMMAs
    // (lets the scheduler stagger s_wait_dscnt instead of full drains)
    v16bf afrag = load_frag(&As[cur][arow * KP1 + kb]);
    v16bf bfrag[8];
    #pragma unroll
    for (int t = 0; t < 8; ++t)
      bfrag[t] = load_frag(&Bs[cur][(t * 16 + (lane & 15)) * KP1 + kb]);
    #pragma unroll
    for (int t = 0; t < 8; ++t)
      acc[t] = __builtin_amdgcn_wmma_f32_16x16x32_bf16(
          false, afrag, false, bfrag[t], (short)0, acc[t], false, false);

    // drain prefetch registers into the other buffer
    if (kt + 1 < KT) {
      uint4* ad = (uint4*)&As[nxt][srow * KP1 + sseg * 16];
      uint4* bd = (uint4*)&Bs[nxt][srow * KP1 + sseg * 16];
      ad[0] = *(uint4*)&xd[0]; ad[1] = *(uint4*)&xd[4];
      bd[0] = *(uint4*)&wd[0]; bd[1] = *(uint4*)&wd[4];
    }
    __syncthreads();
  }

  // write down-projection as bf16 (feeds stage-2 A fragments directly)
  #pragma unroll
  for (int t = 0; t < 8; ++t) {
    const int n = t * 16 + (lane & 15);
    #pragma unroll
    for (int v = 0; v < 8; ++v) {
      const int m = mtile * 128 + wid * 16 + v + ((lane >= 16) ? 8 : 0);
      down[(size_t)m * LORA_R + n] = (__bf16)acc[t][v];
    }
  }
}

// ---- stage 2: out[b, s, o] = SCALE * down[b, s, :] . graph_rep[b, :, o] ---
// grid = (OUT/128, SEQ/128, BSZ), block = 256. Whole K=128 staged once.

__global__ __launch_bounds__(256) void lora_up_wmma(
    const __bf16* __restrict__ down, const float* __restrict__ g,
    float* __restrict__ out)
{
  __shared__ unsigned short As[128 * KP2];  // down tile [m][r] bf16
  __shared__ unsigned short Bs[128 * KP2];  // graph_rep tile [o][r] bf16

  const int tid   = threadIdx.x;
  const int lane  = tid & 31;
  const int wid   = tid >> 5;
  const int ntile = blockIdx.x;
  const int mtile = blockIdx.y;
  const int b     = blockIdx.z;

  // stage A: pure bf16 copy, 128B per thread -> async DMA to LDS if available
  {
    const int row = tid >> 1, seg = tid & 1;
    const __bf16* src = down +
        ((size_t)(b * SEQ + mtile * 128 + row)) * LORA_R + seg * 64;
    unsigned short* dst = &As[row * KP2 + seg * 64];
#if HAVE_ASYNC_LDS
    #pragma unroll
    for (int i = 0; i < 8; ++i)
      __builtin_amdgcn_global_load_async_to_lds_b128(
          (AS1 i32x4*)(src + i * 8), (AS3 i32x4*)(dst + i * 8), 0, 0);
#else
    const uint4* s4 = (const uint4*)src;
    uint4* d4 = (uint4*)dst;
    #pragma unroll
    for (int i = 0; i < 8; ++i) d4[i] = s4[i];
#endif
  }
  // stage B: transpose + hw fp32->bf16. Coalesced along o; packed r-pairs.
  {
    const float* gb = g + (size_t)b * LORA_R * OUT_DIM + (size_t)ntile * 128;
    #pragma unroll 4
    for (int i = 0; i < 32; ++i) {
      const int idx = tid + i * 256;        // 0..8191
      const int o   = idx & 127;
      const int r   = (idx >> 7) * 2;       // 0,2,..,126
      float v0 = gb[(size_t)r       * OUT_DIM + o];
      float v1 = gb[(size_t)(r + 1) * OUT_DIM + o];
      *(unsigned*)&Bs[o * KP2 + r] = pack2_bf16(v0, v1);
    }
  }
  wait_async_then_barrier();

  v8f acc[8] = {};
  const int arow = wid * 16 + (lane & 15);
  const int kb   = (lane < 16) ? 0 : 8;

  #pragma unroll
  for (int kt = 0; kt < LORA_R / 32; ++kt) {     // 4 K-steps
    const int k0 = kt * 32 + kb;
    v16bf afrag = load_frag(&As[arow * KP2 + k0]);
    v16bf bfrag[8];
    #pragma unroll
    for (int t = 0; t < 8; ++t)
      bfrag[t] = load_frag(&Bs[(t * 16 + (lane & 15)) * KP2 + k0]);
    #pragma unroll
    for (int t = 0; t < 8; ++t)
      acc[t] = __builtin_amdgcn_wmma_f32_16x16x32_bf16(
          false, afrag, false, bfrag[t], (short)0, acc[t], false, false);
  }

  float* ob = out + ((size_t)b * SEQ + mtile * 128) * OUT_DIM + (size_t)ntile * 128;
  #pragma unroll
  for (int t = 0; t < 8; ++t) {
    const int n = t * 16 + (lane & 15);
    #pragma unroll
    for (int v = 0; v < 8; ++v) {
      const int m = wid * 16 + v + ((lane >= 16) ? 8 : 0);
      ob[(size_t)m * OUT_DIM + n] = acc[t][v] * SCALE;
    }
  }
}

// ---- host glue ------------------------------------------------------------

extern "C" void kernel_launch(void* const* d_in, const int* in_sizes, int n_in,
                              void* d_out, int out_size, void* d_ws, size_t ws_size,
                              hipStream_t stream) {
  const float* graph_rep = (const float*)d_in[0];  // [8,128,4096]
  const float* x         = (const float*)d_in[1];  // [8,2048,4096]
  const float* W_down    = (const float*)d_in[2];  // [128,4096]
  float* out = (float*)d_out;                      // [8,2048,4096]
  __bf16* down = (__bf16*)d_ws;                    // [16384,128] bf16 = 4 MB scratch

  lora_down_wmma<<<dim3(M_TOTAL / 128), 256, 0, stream>>>(x, W_down, down);
  lora_up_wmma<<<dim3(OUT_DIM / 128, SEQ / 128, BSZ), 256, 0, stream>>>(down, graph_rep, out);
}
